// PAB_80393197846618
// MI455X (gfx1250) — compile-verified, matrix operand order checked by reading, not run on hardware
//
#include <hip/hip_runtime.h>

// Fused Parallax-Attention (PAM) for MI455X / gfx1250.
// One workgroup (256 threads = 8 wave32) per (batch, row, direction).
// Pipeline per block, all in LDS:
//   1) load src row (float4) -> Xt (w,c) f16
//   2) conv1x1+BN (folded) via WMMA -> Qt (w,c) f16  [log2e/64 folded in]
//   3) load tgt row -> Xt (w,c) and V (c,w) f16
//   4) conv1x1+BN via WMMA -> Kt (w,c) f16
//   5) flash-style attention: S^T = K^T Q via WMMA, online softmax in exp2
//      domain (per-lane + one shfl_xor(16)), O^T += V * P via WMMA.

typedef __attribute__((ext_vector_type(16))) _Float16 v16h;
typedef __attribute__((ext_vector_type(8)))  _Float16 v8h;
typedef __attribute__((ext_vector_type(4)))  _Float16 v4h;
typedef __attribute__((ext_vector_type(8)))  float    v8f;

#define NB 8
#define NC 64
#define NH 128
#define NW 256
#define HWSZ (NH * NW)
#define BN_EPS 1e-5f
#define LOG2E 1.44269504088896340736f

// LDS layout (in halfs). All strides multiples of 8 halfs (16B) for b128 ds ops.
#define XT_OFF 0
#define XT_STR 72
#define QT_OFF (256 * 72)
#define KT_OFF (2 * 256 * 72)
#define V_OFF  (3 * 256 * 72)
#define V_STR  264
#define SMEM_HALFS (3 * 256 * 72 + 64 * 264)   // 72192 halfs
#define SMEM_BYTES (SMEM_HALFS * 2)            // 144384 B (<= 320KB/WGP)

__device__ __forceinline__ v8f wmma16(v16h a, v16h b, v8f c) {
  // D = A(16x32 f16) * B(32x16 f16) + C(16x16 f32)
  return __builtin_amdgcn_wmma_f32_16x16x32_f16(false, a, false, b,
                                                (short)0, c, false, false);
}

// Load one 16x32 f16 A/B operand half-row for this lane:
// elements 0..7 = p[0..7], elements 8..15 = p[16..23]  (caller adds the
// lane<16 ? 0 : 8 half-wave K offset).
__device__ __forceinline__ v16h ld_ab(const _Float16* p) {
  v8h lo = *(const v8h*)(p);
  v8h hi = *(const v8h*)(p + 16);
  v16h r;
#pragma unroll
  for (int i = 0; i < 8; ++i) { r[i] = lo[i]; r[8 + i] = hi[i]; }
  return r;
}

// ---------------- setup: fold BN into weights ----------------
// ws layout: halfs [0..4095]=Wq' (x log2e/64), [4096..8191]=Wk';
// bytes 16384: float Bq'[64], Bk'[64].
__global__ void fold_weights(const float* __restrict__ Wq, const float* __restrict__ bq,
                             const float* __restrict__ gq, const float* __restrict__ betaq,
                             const float* __restrict__ mq, const float* __restrict__ vq,
                             const float* __restrict__ Wk, const float* __restrict__ bk,
                             const float* __restrict__ gk, const float* __restrict__ betak,
                             const float* __restrict__ mk, const float* __restrict__ vk,
                             _Float16* __restrict__ wsW, float* __restrict__ wsB) {
  const float qscale = LOG2E / 64.0f;   // cost/64, softmax computed with exp2
  int t = threadIdx.x;
  for (int i = t; i < 4096; i += 256) {
    int o = i >> 6;
    float sq = gq[o] * rsqrtf(vq[o] + BN_EPS);
    float sk = gk[o] * rsqrtf(vk[o] + BN_EPS);
    wsW[i]        = (_Float16)(Wq[i] * sq * qscale);
    wsW[4096 + i] = (_Float16)(Wk[i] * sk);
  }
  if (t < 64) {
    float sq = gq[t] * rsqrtf(vq[t] + BN_EPS);
    float sk = gk[t] * rsqrtf(vk[t] + BN_EPS);
    wsB[t]      = (sq * bq[t] + betaq[t] - mq[t] * sq) * qscale;
    wsB[64 + t] = (sk * bk[t] + betak[t] - mk[t] * sk);
  }
}

// Conv 1x1 over one row: Y(64 out x 256 px) = W'(64x64) X + b', X in Xt (w,c) LDS.
__device__ __forceinline__ void conv_row(_Float16* sm, const _Float16* __restrict__ Wf,
                                         const float* __restrict__ Bf, int dst_off,
                                         int wvs, int hl, int k0) {
  for (int ii = 0; ii < 2; ++ii) {                 // uniform trip: no exec masking
    const int pt = wvs + ii * 8;                   // pixel tile (wave-scalar base)
    const _Float16* xrow = sm + XT_OFF + (pt * 16 + hl) * XT_STR;
    v16h xb0 = ld_ab(xrow + k0);                   // in-ch 0..31  (B operand)
    v16h xb1 = ld_ab(xrow + 32 + k0);              // in-ch 32..63
#pragma unroll
    for (int t = 0; t < 4; ++t) {                  // 4 out-channel tiles
      const _Float16* wrow = Wf + (t * 16 + hl) * 64;
      v16h a0 = ld_ab(wrow + k0);
      v16h a1 = ld_ab(wrow + 32 + k0);
      v8f acc;
#pragma unroll
      for (int k = 0; k < 8; ++k) acc[k] = Bf[t * 16 + k0 + k];
      acc = wmma16(a0, xb0, acc);
      acc = wmma16(a1, xb1, acc);
      v8h hv;
#pragma unroll
      for (int k = 0; k < 8; ++k) hv[k] = (_Float16)acc[k];
      *(v8h*)(sm + dst_off + (pt * 16 + hl) * XT_STR + t * 16 + k0) = hv;
    }
  }
}

__global__ __launch_bounds__(256) void pam_fused(const float* __restrict__ fea_left,
                                                 const float* __restrict__ fea_right,
                                                 const _Float16* __restrict__ wsW,
                                                 const float* __restrict__ wsB,
                                                 float* __restrict__ out) {
  extern __shared__ _Float16 sm[];
  const int bi  = blockIdx.x;
  const int dir = bi >> 10;                 // 1024 = NB*NH blocks per direction
  const int r   = bi & 1023;
  const int b   = r >> 7;
  const int h   = r & 127;
  const float* src = dir ? fea_right : fea_left;   // queries from src
  const float* tgt = dir ? fea_left  : fea_right;  // keys/values from tgt
  const size_t rowoff = (size_t)b * NC * HWSZ + (size_t)h * NW;

  const int tid  = threadIdx.x;
  const int lane = tid & 31;
  const int wvs  = __builtin_amdgcn_readfirstlane(tid >> 5);  // wave id, scalar
  const int hl   = lane & 15;
  const int k0   = (lane < 16) ? 0 : 8;     // ISA half-wave K offset

  // ---- pass 1: src row -> Xt (w,c) f16, float4-vectorized ----
  {
    const float4* sp = (const float4*)(src + rowoff);
    for (int i = tid; i < NC * NW / 4; i += 256) {
      int c = i >> 6, w = (i & 63) * 4;
      float4 v = sp[c * (HWSZ / 4) + (i & 63)];
      sm[XT_OFF + (w + 0) * XT_STR + c] = (_Float16)v.x;
      sm[XT_OFF + (w + 1) * XT_STR + c] = (_Float16)v.y;
      sm[XT_OFF + (w + 2) * XT_STR + c] = (_Float16)v.z;
      sm[XT_OFF + (w + 3) * XT_STR + c] = (_Float16)v.w;
    }
  }
  __syncthreads();

  // ---- conv Q (log2e/64 folded) -> Qt ----
  conv_row(sm, wsW, wsB, QT_OFF, wvs, hl, k0);
  __syncthreads();

  // ---- pass 2: tgt row -> Xt (w,c) and V (c,w) ----
  {
    const float4* tp = (const float4*)(tgt + rowoff);
    for (int i = tid; i < NC * NW / 4; i += 256) {
      int c = i >> 6, w = (i & 63) * 4;
      float4 v = tp[c * (HWSZ / 4) + (i & 63)];
      v4h hv4 = {(_Float16)v.x, (_Float16)v.y, (_Float16)v.z, (_Float16)v.w};
      sm[XT_OFF + (w + 0) * XT_STR + c] = hv4[0];
      sm[XT_OFF + (w + 1) * XT_STR + c] = hv4[1];
      sm[XT_OFF + (w + 2) * XT_STR + c] = hv4[2];
      sm[XT_OFF + (w + 3) * XT_STR + c] = hv4[3];
      *(v4h*)(sm + V_OFF + c * V_STR + w) = hv4;
    }
  }
  __syncthreads();

  // ---- conv K -> Kt ----
  conv_row(sm, wsW + 4096, wsB + 64, KT_OFF, wvs, hl, k0);
  __syncthreads();

  // ---- attention: per wave, 2 query tiles of 16 queries ----
  for (int ii = 0; ii < 2; ++ii) {
    const int qt = wvs + ii * 8;
    const _Float16* qrow = sm + QT_OFF + (qt * 16 + hl) * XT_STR;
    v16h qb0 = ld_ab(qrow + k0);        // Q B-operand, ch 0..31
    v16h qb1 = ld_ab(qrow + 32 + k0);   // ch 32..63

    v8f o[4];
#pragma unroll
    for (int t = 0; t < 4; ++t)
#pragma unroll
      for (int k = 0; k < 8; ++k) o[t][k] = 0.0f;
    float m = -3.0e38f, l = 0.0f;

    for (int kc = 0; kc < 8; ++kc) {    // 8 key chunks of 32 keys
      // K^T A-operands: two key tiles (keys kc*32+0..15 and +16..31)
      const _Float16* kr0 = sm + KT_OFF + (kc * 32 + hl) * XT_STR;
      const _Float16* kr1 = kr0 + 16 * XT_STR;
      v16h ka00 = ld_ab(kr0 + k0), ka01 = ld_ab(kr0 + 32 + k0);
      v16h ka10 = ld_ab(kr1 + k0), ka11 = ld_ab(kr1 + 32 + k0);

      v8f s0, s1;
#pragma unroll
      for (int k = 0; k < 8; ++k) { s0[k] = 0.0f; s1[k] = 0.0f; }
      s0 = wmma16(ka00, qb0, s0); s0 = wmma16(ka01, qb1, s0);
      s1 = wmma16(ka10, qb0, s1); s1 = wmma16(ka11, qb1, s1);

      // online softmax (exp2 domain; log2e folded into Q) per query lane pair
      float cm = s0[0];
#pragma unroll
      for (int k = 1; k < 8; ++k) cm = fmaxf(cm, s0[k]);
#pragma unroll
      for (int k = 0; k < 8; ++k) cm = fmaxf(cm, s1[k]);
      cm = fmaxf(cm, __shfl_xor(cm, 16, 32));
      float mn   = fmaxf(m, cm);
      float corr = __builtin_amdgcn_exp2f(m - mn);
      v8f p0, p1;
      float cs = 0.0f;
#pragma unroll
      for (int k = 0; k < 8; ++k) { p0[k] = __builtin_amdgcn_exp2f(s0[k] - mn); cs += p0[k]; }
#pragma unroll
      for (int k = 0; k < 8; ++k) { p1[k] = __builtin_amdgcn_exp2f(s1[k] - mn); cs += p1[k]; }
      cs += __shfl_xor(cs, 16, 32);
      l = l * corr + cs;
      m = mn;
#pragma unroll
      for (int t = 0; t < 4; ++t) o[t] = o[t] * corr;

      // P^T C-layout -> B-operand by pure per-lane f32->f16 pair packing
      v16h pb;
#pragma unroll
      for (int e = 0; e < 8; ++e) { pb[e] = (_Float16)p0[e]; pb[8 + e] = (_Float16)p1[e]; }

      // O^T += V(ch tile, 32 keys) * P
#pragma unroll
      for (int ct = 0; ct < 4; ++ct) {
        v16h va = ld_ab(sm + V_OFF + (ct * 16 + hl) * V_STR + kc * 32 + k0);
        o[ct] = wmma16(va, pb, o[ct]);
      }
    }

    // finalize and write out (B,C,H,W) f32; dir selects cxr vs enh half
    float inv = 1.0f / l;
    float* obase = out + (size_t)dir * ((size_t)NB * NC * HWSZ)
                 + (size_t)b * NC * HWSZ + (size_t)h * NW + qt * 16 + hl;
#pragma unroll
    for (int ct = 0; ct < 4; ++ct)
#pragma unroll
      for (int k = 0; k < 8; ++k)
        obase[(size_t)(ct * 16 + k0 + k) * HWSZ] = o[ct][k] * inv;
  }
}

extern "C" void kernel_launch(void* const* d_in, const int* in_sizes, int n_in,
                              void* d_out, int out_size, void* d_ws, size_t ws_size,
                              hipStream_t stream) {
  (void)in_sizes; (void)n_in; (void)out_size; (void)ws_size;
  const float* fl = (const float*)d_in[0];
  const float* fr = (const float*)d_in[1];
  _Float16* wsW = (_Float16*)d_ws;
  float*    wsB = (float*)((char*)d_ws + 16384);

  fold_weights<<<1, 256, 0, stream>>>(
      (const float*)d_in[2],  (const float*)d_in[3],  (const float*)d_in[4],
      (const float*)d_in[5],  (const float*)d_in[6],  (const float*)d_in[7],
      (const float*)d_in[8],  (const float*)d_in[9],  (const float*)d_in[10],
      (const float*)d_in[11], (const float*)d_in[12], (const float*)d_in[13],
      wsW, wsB);

  (void)hipFuncSetAttribute((const void*)pam_fused,
                            hipFuncAttributeMaxDynamicSharedMemorySize, SMEM_BYTES);
  pam_fused<<<dim3(NB * NH * 2), dim3(256), SMEM_BYTES, stream>>>(
      fl, fr, wsW, wsB, (float*)d_out);
}
